// MultiHeadAttentionWrapper_8366596293037
// MI455X (gfx1250) — compile-verified, matrix operand order checked by reading, not run on hardware
//
#include <hip/hip_runtime.h>

// ---------------------------------------------------------------------------
// Problem constants (from reference): B=4, N=1024, D=512, H=8, HD=64, HH=4
// ---------------------------------------------------------------------------
#define BB 4
#define NN 1024
#define DD 512
#define HH_HEADS 8
#define HD 64
#define HSHORT 4

typedef __attribute__((ext_vector_type(16))) __bf16 v16bf;
typedef __attribute__((ext_vector_type(8)))  __bf16 v8bf;
typedef __attribute__((ext_vector_type(8)))  float  v8f;
typedef __attribute__((ext_vector_type(4)))  float  v4f;

// ---------------- bf16 helpers (bit-level, no reliance on __bf16 arith) ----
__device__ __forceinline__ __bf16 f2bf(float f) {
    unsigned u = __float_as_uint(f);
    unsigned r = (u + 0x7FFFu + ((u >> 16) & 1u)) >> 16;
    unsigned short s = (unsigned short)r;
    __bf16 b;
    __builtin_memcpy(&b, &s, 2);
    return b;
}

// ---------------------------------------------------------------------------
// Load a 16x32 bf16 A/B fragment from a row-major source.
// Fragment element e (VGPR e/2, half e%2) of lane L maps to
//   row = L%16,  k = (e<8 ? h*8+e : 16 + h*8 + (e-8)),  h = L/16
// which is two contiguous 8-element (16-byte) chunks per lane.
// ---------------------------------------------------------------------------
__device__ __forceinline__ v16bf load_frag16(const __bf16* base, int ld, int lane) {
    int r = lane & 15;
    int h = lane >> 4;
    const __bf16* p0 = base + (size_t)r * ld + h * 8;
    const __bf16* p1 = p0 + 16;
    union { v16bf v; v8bf half[2]; } u;
    u.half[0] = *(const v8bf*)p0;
    u.half[1] = *(const v8bf*)p1;
    return u.v;
}

// ---------------------------------------------------------------------------
// fp32 -> bf16 elementwise conversion
// ---------------------------------------------------------------------------
__global__ void cvt_f32_bf16(const float* __restrict__ src,
                             __bf16* __restrict__ dst, int n) {
    int i = blockIdx.x * 256 + threadIdx.x;
    if (i < n) dst[i] = f2bf(src[i]);
}

// ---------------------------------------------------------------------------
// Y = X (MxK, bf16 row-major) * W^T (W is NxK bf16 row-major) + bias
// mode 0: store fp32 row-major  Y[m*Nn + n]
// mode 1: store bf16 row-major  Y[m*Nn + n]
// mode 2: store bf16 transposed Y[n*M  + m]
// One 16x64 strip per wave (4 accumulators, A-fragment reused), 4 waves/block.
// ---------------------------------------------------------------------------
__global__ __launch_bounds__(128)
void gemm_bt_wmma(const __bf16* __restrict__ X, const __bf16* __restrict__ W,
                  const float* __restrict__ bias, void* __restrict__ Yv,
                  int M, int Nn, int K, int mode) {
    int wave = threadIdx.x >> 5;
    int lane = threadIdx.x & 31;
    int g = blockIdx.x * 4 + wave;
    int ntn = Nn >> 6;                    // 64-wide N strips
    int total = (M >> 4) * ntn;
    if (g >= total) return;               // wave-uniform: EXEC stays all-ones
    int tm = g / ntn;
    int tn = g % ntn;

    const __bf16* xbase = X + (size_t)(tm * 16) * K;
    const __bf16* wbase = W + (size_t)(tn * 64) * K;

    v8f c0, c1, c2, c3;
#pragma unroll
    for (int i = 0; i < 8; ++i) { c0[i] = 0.0f; c1[i] = 0.0f; c2[i] = 0.0f; c3[i] = 0.0f; }

    for (int kk = 0; kk < K; kk += 32) {
        v16bf a  = load_frag16(xbase + kk, K, lane);
        v16bf b0 = load_frag16(wbase + (size_t)0  * K + kk, K, lane);
        v16bf b1 = load_frag16(wbase + (size_t)16 * K + kk, K, lane);
        v16bf b2 = load_frag16(wbase + (size_t)32 * K + kk, K, lane);
        v16bf b3 = load_frag16(wbase + (size_t)48 * K + kk, K, lane);
        c0 = __builtin_amdgcn_wmma_f32_16x16x32_bf16(false, a, false, b0,
                                                     (short)0, c0, false, false);
        c1 = __builtin_amdgcn_wmma_f32_16x16x32_bf16(false, a, false, b1,
                                                     (short)0, c1, false, false);
        c2 = __builtin_amdgcn_wmma_f32_16x16x32_bf16(false, a, false, b2,
                                                     (short)0, c2, false, false);
        c3 = __builtin_amdgcn_wmma_f32_16x16x32_bf16(false, a, false, b3,
                                                     (short)0, c3, false, false);
    }

    int nloc = lane & 15;
    int hh   = lane >> 4;

#pragma unroll
    for (int sub = 0; sub < 4; ++sub) {
        v8f c = (sub == 0) ? c0 : (sub == 1) ? c1 : (sub == 2) ? c2 : c3;
        int n = tn * 64 + sub * 16 + nloc;
        float bs = bias ? bias[n] : 0.0f;
#pragma unroll
        for (int v = 0; v < 8; ++v) {
            int m = tm * 16 + hh * 8 + v;
            float val = c[v] + bs;
            if (mode == 0) {
                ((float*)Yv)[(size_t)m * Nn + n] = val;
            } else if (mode == 1) {
                ((__bf16*)Yv)[(size_t)m * Nn + n] = f2bf(val);
            } else {
                ((__bf16*)Yv)[(size_t)n * M + m] = f2bf(val);
            }
        }
    }
}

// ---------------------------------------------------------------------------
// Fused attention: one block = (b, h, 16-query strip).
//  Phase 1: S = scale * Q Kh^T  (WMMA, K=64 via 2 steps), short heads get
//           gamma^relu(sph - hop) mask. S -> LDS (16 x 1024 fp32).
//  Phase 2: row softmax in LDS; normalized weights -> global (d_out region 2);
//           S keeps unnormalized exp values.
//  Phase 3: out = (exp S) * V via WMMA (A from LDS, B from transposed V),
//           rows scaled by 1/rowsum; merged bf16 -> workspace.
// ---------------------------------------------------------------------------
__global__ __launch_bounds__(128)
void attn_wmma(const __bf16* __restrict__ Qp, const __bf16* __restrict__ Kp,
               const __bf16* __restrict__ Vt, const float* __restrict__ sph,
               const float* __restrict__ hop, const float* __restrict__ sgam,
               float* __restrict__ weights, __bf16* __restrict__ merged) {
    __shared__ float S[16][NN];
    __shared__ float red[16][8];
    __shared__ float rowmax[16];
    __shared__ float rowinv[16];

    int qt = blockIdx.x & 63;                 // query tile (N/16 = 64)
    int h  = (blockIdx.x >> 6) & 7;           // head
    int b  = blockIdx.x >> 9;                 // batch

    int wave = threadIdx.x >> 5;
    int lane = threadIdx.x & 31;
    int nloc = lane & 15;
    int hh   = lane >> 4;

    const float scale = 0.125f;               // 1/sqrt(64)
    bool is_short = (h < HSHORT);
    float l2g = 0.0f, hopv = 0.0f;
    if (is_short) {
        float g0    = sgam[h];
        float gamma = 1.0f / (1.0f + __expf(-g0));
        l2g  = __log2f(gamma);
        hopv = hop[h];
    }

    // ---- Phase 1: scores ----
    const __bf16* qbase = Qp + ((size_t)(b * NN + qt * 16)) * DD + h * HD;
    v16bf aq0 = load_frag16(qbase,      DD, lane);
    v16bf aq1 = load_frag16(qbase + 32, DD, lane);

    for (int t = wave; t < NN / 16; t += 4) {
        const __bf16* kbase = Kp + ((size_t)(b * NN + t * 16)) * DD + h * HD;
        v16bf bk0 = load_frag16(kbase,      DD, lane);
        v16bf bk1 = load_frag16(kbase + 32, DD, lane);
        v8f c;
#pragma unroll
        for (int i = 0; i < 8; ++i) c[i] = 0.0f;
        c = __builtin_amdgcn_wmma_f32_16x16x32_bf16(false, aq0, false, bk0,
                                                    (short)0, c, false, false);
        c = __builtin_amdgcn_wmma_f32_16x16x32_bf16(false, aq1, false, bk1,
                                                    (short)0, c, false, false);

        const float* sphbase =
            sph + ((size_t)(b * NN + qt * 16)) * NN + t * 16 + nloc;
#pragma unroll
        for (int v = 0; v < 8; ++v) {
            int m = hh * 8 + v;               // local query row
            float s = c[v] * scale;
            if (is_short) {
                float d = sphbase[(size_t)m * NN] - hopv;
                d = d > 0.0f ? d : 0.0f;
                s *= exp2f(l2g * d);
            }
            S[m][t * 16 + nloc] = s;
        }
    }
    __syncthreads();

    // ---- Phase 2: softmax over 1024 columns per row ----
    int r  = threadIdx.x & 15;
    int ck = threadIdx.x >> 4;                // 8 chunks x 128 cols
    float mx = -1e30f;
    for (int j = 0; j < 128; ++j) mx = fmaxf(mx, S[r][ck * 128 + j]);
    red[r][ck] = mx;
    __syncthreads();
    if (threadIdx.x < 16) {
        float m2 = red[threadIdx.x][0];
#pragma unroll
        for (int j = 1; j < 8; ++j) m2 = fmaxf(m2, red[threadIdx.x][j]);
        rowmax[threadIdx.x] = m2;
    }
    __syncthreads();
    float rm = rowmax[r];
    float sum = 0.0f;
    for (int j = 0; j < 128; ++j) {
        float e = __expf(S[r][ck * 128 + j] - rm);
        S[r][ck * 128 + j] = e;
        sum += e;
    }
    red[r][ck] = sum;
    __syncthreads();
    if (threadIdx.x < 16) {
        float s2 = 0.0f;
#pragma unroll
        for (int j = 0; j < 8; ++j) s2 += red[threadIdx.x][j];
        rowinv[threadIdx.x] = 1.0f / s2;
    }
    __syncthreads();
    float inv = rowinv[r];
    float* wrow =
        weights + (((size_t)(b * HH_HEADS + h) * NN) + qt * 16 + r) * NN;
    for (int j = 0; j < 128; ++j)
        wrow[ck * 128 + j] = S[r][ck * 128 + j] * inv;

    // ---- Phase 3: out = (exp S) @ V, rows scaled by 1/rowsum ----
    // wave handles output columns hd = wave*16 .. wave*16+15
    v8f acc;
#pragma unroll
    for (int i = 0; i < 8; ++i) acc[i] = 0.0f;

    const __bf16* vbase =
        Vt + ((size_t)(h * HD + wave * 16 + nloc)) * (BB * NN) + b * NN;
    int qrow = nloc;                          // A-frag: lane row = query
    for (int kk = 0; kk < NN; kk += 32) {
        // A fragment from LDS exp-values: 4x ds_load_b128 + cvt
        v4f s0 = *(const v4f*)&S[qrow][kk + hh * 8];
        v4f s1 = *(const v4f*)&S[qrow][kk + hh * 8 + 4];
        v4f s2 = *(const v4f*)&S[qrow][kk + 16 + hh * 8];
        v4f s3 = *(const v4f*)&S[qrow][kk + 16 + hh * 8 + 4];
        v16bf a;
#pragma unroll
        for (int e = 0; e < 4; ++e) {
            a[e]      = f2bf(s0[e]);
            a[4 + e]  = f2bf(s1[e]);
            a[8 + e]  = f2bf(s2[e]);
            a[12 + e] = f2bf(s3[e]);
        }

        union { v16bf v; v8bf half[2]; } ub;
        ub.half[0] = *(const v8bf*)(vbase + kk + hh * 8);
        ub.half[1] = *(const v8bf*)(vbase + kk + 16 + hh * 8);

        acc = __builtin_amdgcn_wmma_f32_16x16x32_bf16(false, a, false, ub.v,
                                                      (short)0, acc, false, false);
    }

    __bf16* mb = merged + ((size_t)(b * NN + qt * 16)) * DD +
                 h * HD + wave * 16 + nloc;
#pragma unroll
    for (int v = 0; v < 8; ++v) {
        int m = hh * 8 + v;
        mb[(size_t)m * DD] = f2bf(acc[v] * rowinv[m]);
    }
}

// ---------------------------------------------------------------------------
// Host-side launcher
// ---------------------------------------------------------------------------
extern "C" void kernel_launch(void* const* d_in, const int* in_sizes, int n_in,
                              void* d_out, int out_size, void* d_ws, size_t ws_size,
                              hipStream_t stream) {
    (void)in_sizes; (void)n_in; (void)out_size; (void)ws_size;

    const float* q    = (const float*)d_in[0];
    const float* k    = (const float*)d_in[1];
    const float* v    = (const float*)d_in[2];
    const float* sph  = (const float*)d_in[3];
    const float* Wq   = (const float*)d_in[4];
    const float* bq   = (const float*)d_in[5];
    const float* Wk   = (const float*)d_in[6];
    const float* bk   = (const float*)d_in[7];
    const float* Wv   = (const float*)d_in[8];
    const float* bv   = (const float*)d_in[9];
    const float* Wo   = (const float*)d_in[10];
    const float* bo   = (const float*)d_in[11];
    const float* hop  = (const float*)d_in[12];
    const float* sgam = (const float*)d_in[13];

    const size_t TOK  = (size_t)BB * NN;          // 4096 tokens
    const size_t QKV  = TOK * DD;                 // 2,097,152 elements
    const size_t WEL  = (size_t)DD * DD;          // 262,144 elements

    char* ws = (char*)d_ws;
    size_t off = 0;
    auto take = [&](size_t bytes) { char* p = ws + off; off += (bytes + 255) & ~(size_t)255; return p; };

    __bf16* qb   = (__bf16*)take(QKV * 2);
    __bf16* kb   = (__bf16*)take(QKV * 2);
    __bf16* vb   = (__bf16*)take(QKV * 2);
    __bf16* wqb  = (__bf16*)take(WEL * 2);
    __bf16* wkb  = (__bf16*)take(WEL * 2);
    __bf16* wvb  = (__bf16*)take(WEL * 2);
    __bf16* wob  = (__bf16*)take(WEL * 2);
    __bf16* pQ   = (__bf16*)take(QKV * 2);        // projected Q, [token][e] bf16
    __bf16* pK   = (__bf16*)take(QKV * 2);        // projected K, [token][e] bf16
    __bf16* pVt  = (__bf16*)take(QKV * 2);        // projected V, transposed [e][token]
    __bf16* merg = (__bf16*)take(QKV * 2);        // merged head outputs [token][e]

    float* out_f   = (float*)d_out;                         // (B,N,D)
    float* weights = out_f + QKV;                           // (B,H,N,N)

    // 1) fp32 -> bf16 conversions
    {
        int n1 = (int)QKV, n2 = (int)WEL;
        int g1 = (n1 + 255) / 256, g2 = (n2 + 255) / 256;
        cvt_f32_bf16<<<g1, 256, 0, stream>>>(q,  qb,  n1);
        cvt_f32_bf16<<<g1, 256, 0, stream>>>(k,  kb,  n1);
        cvt_f32_bf16<<<g1, 256, 0, stream>>>(v,  vb,  n1);
        cvt_f32_bf16<<<g2, 256, 0, stream>>>(Wq, wqb, n2);
        cvt_f32_bf16<<<g2, 256, 0, stream>>>(Wk, wkb, n2);
        cvt_f32_bf16<<<g2, 256, 0, stream>>>(Wv, wvb, n2);
        cvt_f32_bf16<<<g2, 256, 0, stream>>>(Wo, wob, n2);
    }

    // 2) projections via WMMA GEMM:  (4096x512) @ (512x512)^T
    {
        int tiles  = ((int)TOK / 16) * (DD / 64);   // 2048 strips
        int blocks = tiles / 4;                     // 512
        gemm_bt_wmma<<<blocks, 128, 0, stream>>>(qb, wqb, bq, pQ,
                                                 (int)TOK, DD, DD, 1);
        gemm_bt_wmma<<<blocks, 128, 0, stream>>>(kb, wkb, bk, pK,
                                                 (int)TOK, DD, DD, 1);
        gemm_bt_wmma<<<blocks, 128, 0, stream>>>(vb, wvb, bv, pVt,
                                                 (int)TOK, DD, DD, 2);
    }

    // 3) fused attention (scores + softmax + weights-out + PV)
    {
        int blocks = BB * HH_HEADS * (NN / 16);     // 2048
        attn_wmma<<<blocks, 128, 0, stream>>>(pQ, pK, pVt, sph, hop, sgam,
                                              weights, merg);
    }

    // 4) output projection: out = merged @ Wo^T + bo  (fp32 store)
    {
        int tiles  = ((int)TOK / 16) * (DD / 64);
        int blocks = tiles / 4;
        gemm_bt_wmma<<<blocks, 128, 0, stream>>>(merg, wob, bo, out_f,
                                                 (int)TOK, DD, DD, 0);
    }
}
